// LearnCutV1_58291296141644
// MI455X (gfx1250) — compile-verified, matrix operand order checked by reading, not run on hardware
//
#include <hip/hip_runtime.h>
#include <stdint.h>

// ---------------- CDNA5 WMMA plumbing ----------------
typedef __attribute__((ext_vector_type(16))) __bf16 v16bf;
typedef __attribute__((ext_vector_type(8)))  float  v8f;

#define B_     2
#define C_     256
#define H_     128
#define K_     64
#define KG_    16
#define NSP    1024
#define MC     128
#define EPS_   1e-4f
#define TAU_   0.1f
#define GAMMA_ 64.0f

__device__ inline unsigned short f2bf(float f) {
  union { float f; unsigned u; } v; v.f = f;
  unsigned u = v.u;
  unsigned r = (u + 0x7FFFu + ((u >> 16) & 1u)) >> 16;
  return (unsigned short)r;
}
__device__ inline float bf2f(unsigned short h) {
  union { unsigned u; float f; } v; v.u = ((unsigned)h) << 16;
  return v.f;
}

struct U4 { unsigned a, b, c, d; };
union FragCast { v16bf v; U4 q[2]; };

// Load a 16x32 bf16 fragment (A-layout) from a row-major matrix with leading dim ld.
// lane L holds row i0+(L%16); halves j<8 -> k=base+j, j>=8 -> k=base+8+j, base=k0+(L>=16?8:0)
__device__ inline v16bf load_frag(const unsigned short* base, int ld, int i0, int k0) {
  int lane = threadIdx.x & 31;
  const unsigned short* p = base + (size_t)(i0 + (lane & 15)) * ld + k0 + ((lane & 16) ? 8 : 0);
  FragCast f;
  f.q[0] = *(const U4*)p;
  f.q[1] = *(const U4*)(p + 16);
  return f.v;
}

__device__ inline v8f wmma_bf(v16bf a, v16bf b, v8f c) {
  return __builtin_amdgcn_wmma_f32_16x16x32_bf16(false, a, false, b, (short)0, c, false, false);
}
__device__ inline v8f v8f_zero() { v8f z = {0.f,0.f,0.f,0.f,0.f,0.f,0.f,0.f}; return z; }
__device__ inline void wait_ds() { asm volatile("s_wait_dscnt 0" ::: "memory"); }

// ---------------- prep kernels ----------------
__global__ void k_transpose_bf16(const float* src, unsigned short* dst, int R, int Cc) {
  int i = blockIdx.x * blockDim.x + threadIdx.x;
  if (i >= R * Cc) return;
  int r = i / Cc, c = i % Cc;
  dst[(size_t)c * R + r] = f2bf(src[i]);
}

// one 256-thread block per row; computes row l2-norm and emits converted copies
__global__ void k_rownorm(const float* fea, int N,
                          float* nf_f32, unsigned short* nf_bf,
                          unsigned short* x_bf, unsigned short* xt_bf) {
  int row = blockIdx.x;           // b*N + n
  int b = row / N, n = row % N;
  const float* src = fea + (size_t)row * C_;
  __shared__ float red[256];
  float v = src[threadIdx.x];
  red[threadIdx.x] = v * v;
  __syncthreads();
  for (int s = 128; s > 0; s >>= 1) {
    if (threadIdx.x < s) red[threadIdx.x] += red[threadIdx.x + s];
    __syncthreads();
  }
  float inv = 1.0f / fmaxf(sqrtf(red[0]), 1e-12f);
  float nv = v * inv;
  size_t idx = (size_t)row * C_ + threadIdx.x;
  if (nf_f32) nf_f32[idx] = nv;
  if (nf_bf)  nf_bf[idx]  = f2bf(nv);
  if (x_bf)   x_bf[idx]   = f2bf(v);
  if (xt_bf)  xt_bf[((size_t)b * C_ + threadIdx.x) * N + n] = f2bf(v);
}

// ---------------- affinity degree (row sums of A) via WMMA gram ----------------
// Workgroup-cooperative: 16-row column panels of nf staged in LDS once per block;
// the wave's A-row panel is held in registers (afr) across the whole m0 sweep.
__global__ void k_deg(const unsigned short* nf_bf, const float* xyz, float* deg,
                      int N, float r2) {
  __shared__ __align__(16) unsigned short btile[16 * C_];   // 8KB shared column panel
  __shared__ float cxyzs[16][3];
  __shared__ float rxyz[8][16][3];
  __shared__ float red[8][16];
  int wv = threadIdx.x >> 5, lane = threadIdx.x & 31;
  int gw = blockIdx.x * 8 + wv;
  int rb = N >> 4;
  int b = gw / rb;
  int r0 = (gw % rb) << 4;
  const unsigned short* nfb = nf_bf + (size_t)b * N * C_;
  const float* xb = xyz + (size_t)b * N * 3;
  if (lane < 16) {
    rxyz[wv][lane][0] = xb[(r0 + lane) * 3 + 0];
    rxyz[wv][lane][1] = xb[(r0 + lane) * 3 + 1];
    rxyz[wv][lane][2] = xb[(r0 + lane) * 3 + 2];
    red[wv][lane] = 0.f;
  }
  v16bf afr[8];
#pragma unroll
  for (int j = 0; j < 8; j++) afr[j] = load_frag(nfb, C_, r0, j * 32);
  float rs[8];
  for (int i = 0; i < 8; i++) rs[i] = 0.f;
  int colb = lane & 15, rsh = (lane & 16) ? 8 : 0;
  int srow = threadIdx.x >> 4, sseg = threadIdx.x & 15;     // staging: 16 rows x 16 segs
  for (int m0 = 0; m0 < N; m0 += 16) {
    __syncthreads();
    {
      const U4* src = (const U4*)(nfb + (size_t)(m0 + srow) * C_ + sseg * 16);
      U4* dst = (U4*)(&btile[srow * C_ + sseg * 16]);
      dst[0] = src[0];
      dst[1] = src[1];
      if (m0 + 16 < N)
        __builtin_prefetch(nfb + (size_t)(m0 + 16 + srow) * C_ + sseg * 16, 0, 1);
    }
    if (threadIdx.x < 48)
      cxyzs[threadIdx.x / 3][threadIdx.x % 3] =
          xb[(size_t)(m0 + threadIdx.x / 3) * 3 + threadIdx.x % 3];
    __syncthreads();
    v8f acc = v8f_zero();
#pragma unroll
    for (int j = 0; j < 8; j++) {
      v16bf bb = load_frag(btile, C_, 0, j * 32);
      acc = wmma_bf(afr[j], bb, acc);
    }
    int col = m0 + colb;
    float cx = cxyzs[colb][0], cy = cxyzs[colb][1], cz = cxyzs[colb][2];
    for (int i = 0; i < 8; i++) {
      int row = r0 + i + rsh;
      float dx = rxyz[wv][i+rsh][0]-cx, dy = rxyz[wv][i+rsh][1]-cy, dz = rxyz[wv][i+rsh][2]-cz;
      float d2x = dx*dx + dy*dy + dz*dz;
      float d2f = fmaxf(2.f - 2.f * acc[i], 0.f);   // nf rows are unit
      float A = __expf(-GAMMA_ * d2f / (float)C_);
      A = (d2x < r2 && row != col) ? A : 0.f;
      rs[i] += A;
    }
  }
  for (int i = 0; i < 8; i++) atomicAdd(&red[wv][i + rsh], rs[i]);
  __syncthreads();
  if (lane < 16) deg[(size_t)b * N + r0 + lane] = red[wv][lane];
}

__global__ void k_rsqrt(const float* deg, float* dinv, int n) {
  int i = blockIdx.x * blockDim.x + threadIdx.x;
  if (i < n) dinv[i] = rsqrtf(deg[i] + EPS_);
}

// ---------------- fused M = D^-1/2 A D^-1/2 X ----------------
// Recomputes A tiles (WMMA gram: register-resident A panel x LDS-staged panel),
// stages na tile in LDS, multiplies against the LDS-staged Xt panel.
__global__ void k_naX(const unsigned short* nf_bf, const unsigned short* xt_bf,
                      const float* xyz, const float* dinv,
                      unsigned short* m_bf, int N, float r2) {
  __shared__ __align__(16) unsigned short btile[32 * C_];   // 16KB nf rows m0..m0+31
  __shared__ __align__(16) unsigned short xtile[C_ * 32];   // 16KB Xt cols m0..m0+31
  __shared__ float cxyzs[32][3];
  __shared__ float cdins[32];
  __shared__ float rxyz[8][16][3];
  __shared__ float rdin[8][16];
  __shared__ __align__(16) unsigned short atile[8][16 * 32];
  int wv = threadIdx.x >> 5, lane = threadIdx.x & 31;
  int gw = blockIdx.x * 8 + wv;
  int rb = N >> 4;
  int b = gw / rb;
  int r0 = (gw % rb) << 4;
  const unsigned short* nfb = nf_bf + (size_t)b * N * C_;
  const unsigned short* xtb = xt_bf + (size_t)b * C_ * N;
  const float* xb = xyz + (size_t)b * N * 3;
  const float* db = dinv + (size_t)b * N;
  if (lane < 16) {
    rxyz[wv][lane][0] = xb[(r0+lane)*3+0];
    rxyz[wv][lane][1] = xb[(r0+lane)*3+1];
    rxyz[wv][lane][2] = xb[(r0+lane)*3+2];
    rdin[wv][lane] = db[r0 + lane];
  }
  v16bf afr[8];
#pragma unroll
  for (int j = 0; j < 8; j++) afr[j] = load_frag(nfb, C_, r0, j * 32);
  v8f cacc[16];
  for (int f = 0; f < 16; f++) cacc[f] = v8f_zero();
  int colb = lane & 15, rsh = (lane & 16) ? 8 : 0;
  int srow = threadIdx.x >> 3, sseg = threadIdx.x & 7;      // staging: 32 rows x 8 segs
  for (int m0 = 0; m0 < N; m0 += 32) {
    __syncthreads();
    {
      const U4* src = (const U4*)(nfb + (size_t)(m0 + srow) * C_ + sseg * 32);
      U4* dst = (U4*)(&btile[srow * C_ + sseg * 32]);
      dst[0]=src[0]; dst[1]=src[1]; dst[2]=src[2]; dst[3]=src[3];
    }
    {
      const U4* src = (const U4*)(xtb + (size_t)threadIdx.x * N + m0);
      U4* dst = (U4*)(&xtile[threadIdx.x * 32]);
      dst[0]=src[0]; dst[1]=src[1]; dst[2]=src[2]; dst[3]=src[3];
      if (m0 + 32 < N)
        __builtin_prefetch(xtb + (size_t)threadIdx.x * N + m0 + 32, 0, 1);
    }
    if (threadIdx.x < 96)
      cxyzs[threadIdx.x/3][threadIdx.x%3] =
          xb[(size_t)(m0 + threadIdx.x/3) * 3 + threadIdx.x%3];
    if (threadIdx.x < 32) cdins[threadIdx.x] = db[m0 + threadIdx.x];
    __syncthreads();
    for (int t = 0; t < 2; t++) {
      v8f acc = v8f_zero();
#pragma unroll
      for (int j = 0; j < 8; j++) {
        v16bf bb = load_frag(btile, C_, t * 16, j * 32);
        acc = wmma_bf(afr[j], bb, acc);
      }
      int cl = t * 16 + colb;
      int col = m0 + cl;
      float cx = cxyzs[cl][0], cy = cxyzs[cl][1], cz = cxyzs[cl][2];
      float dc = cdins[cl];
      for (int i = 0; i < 8; i++) {
        int row = r0 + i + rsh;
        float dx = rxyz[wv][i+rsh][0]-cx, dy = rxyz[wv][i+rsh][1]-cy, dz = rxyz[wv][i+rsh][2]-cz;
        float d2x = dx*dx + dy*dy + dz*dz;
        float d2f = fmaxf(2.f - 2.f * acc[i], 0.f);
        float A = __expf(-GAMMA_ * d2f / (float)C_);
        A = (d2x < r2 && row != col) ? A : 0.f;
        atile[wv][(i + rsh) * 32 + cl] = f2bf(A * rdin[wv][i+rsh] * dc);
      }
    }
    wait_ds();
    v16bf a = load_frag(&atile[wv][0], 32, 0, 0);
    for (int f = 0; f < 16; f++) {
      v16bf bb = load_frag(xtile, 32, f * 16, 0);
      cacc[f] = wmma_bf(a, bb, cacc[f]);
    }
  }
  unsigned short* mb = m_bf + (size_t)b * N * C_;
  for (int f = 0; f < 16; f++)
    for (int i = 0; i < 8; i++)
      mb[(size_t)(r0 + i + rsh) * C_ + f * 16 + colb] = f2bf(cacc[f][i]);
}

// ---------------- fused GCN layer + temperature softmax ----------------
__global__ void k_mincut(const unsigned short* m_bf, const unsigned short* w1t,
                         const float* b1, const unsigned short* w2t, const float* b2,
                         float* s_f32, unsigned short* s_bf, unsigned short* st_bf,
                         float* colsum, int N) {
  __shared__ __align__(16) unsigned short htile[8][16 * H_];
  __shared__ float stile[8][16 * K_];
  int wv = threadIdx.x >> 5, lane = threadIdx.x & 31;
  int gw = blockIdx.x * 8 + wv;
  int rb = N >> 4;
  int b = gw / rb;
  int r0 = (gw % rb) << 4;
  const unsigned short* mb = m_bf + (size_t)b * N * C_;
  int colb = lane & 15, rsh = (lane & 16) ? 8 : 0;

  v8f hacc[8];
  for (int f = 0; f < 8; f++) hacc[f] = v8f_zero();
  for (int kk = 0; kk < C_; kk += 32) {
    v16bf a = load_frag(mb, C_, r0, kk);
    for (int f = 0; f < 8; f++) {
      v16bf bb = load_frag(w1t, C_, f * 16, kk);
      hacc[f] = wmma_bf(a, bb, hacc[f]);
    }
  }
  for (int f = 0; f < 8; f++) {
    int c = f * 16 + colb;
    float bias = b1[c];
    for (int i = 0; i < 8; i++)
      htile[wv][(i + rsh) * H_ + c] = f2bf(fmaxf(hacc[f][i] + bias, 0.f));
  }
  wait_ds();
  v8f sacc[4];
  for (int f = 0; f < 4; f++) sacc[f] = v8f_zero();
  for (int kk = 0; kk < H_; kk += 32) {
    v16bf a = load_frag(&htile[wv][0], H_, 0, kk);
    for (int f = 0; f < 4; f++) {
      v16bf bb = load_frag(w2t, H_, f * 16, kk);
      sacc[f] = wmma_bf(a, bb, sacc[f]);
    }
  }
  for (int f = 0; f < 4; f++) {
    int c = f * 16 + colb;
    float bias = b2[c];
    for (int i = 0; i < 8; i++)
      stile[wv][(i + rsh) * K_ + c] = (sacc[f][i] + bias) * (1.0f / TAU_);
  }
  wait_ds();
  if (lane < 16) {
    float* row = &stile[wv][lane * K_];
    float mx = row[0];
    for (int k = 1; k < K_; k++) mx = fmaxf(mx, row[k]);
    float sm = 0.f;
    for (int k = 0; k < K_; k++) { float e = __expf(row[k] - mx); row[k] = e; sm += e; }
    float inv = 1.f / sm;
    for (int k = 0; k < K_; k++) row[k] *= inv;
  }
  wait_ds();
  float* sfb = s_f32 + (size_t)b * N * K_;
  unsigned short* sbb = s_bf + (size_t)b * N * K_;
  unsigned short* stb = st_bf + (size_t)b * K_ * N;
  for (int e = lane; e < 16 * K_; e += 32) {
    int rr = e / K_, cc = e % K_;
    float v = stile[wv][rr * K_ + cc];
    int row = r0 + rr;
    sfb[(size_t)row * K_ + cc] = v;
    sbb[(size_t)row * K_ + cc] = f2bf(v);
    stb[(size_t)cc * N + row]  = f2bf(v);
  }
  for (int k = lane; k < K_; k += 32) {
    float cs = 0.f;
    for (int rr = 0; rr < 16; rr++) cs += stile[wv][rr * K_ + k];
    atomicAdd(&colsum[b * K_ + k], cs);
  }
}

// ---------------- c_fea = s^T X / denom (64x256 per batch) ----------------
__global__ void k_cfea(const unsigned short* st_bf, const unsigned short* xt_bf,
                       const float* colsum, float* cfea_cat, unsigned short* cfeaT_bf,
                       int N, int lvloff) {
  int b = blockIdx.x;
  int wv = threadIdx.x >> 5, lane = threadIdx.x & 31;
  const unsigned short* stb = st_bf + (size_t)b * K_ * N;
  const unsigned short* xtb = xt_bf + (size_t)b * C_ * N;
  int colb = lane & 15, rsh = (lane & 16) ? 8 : 0;
  for (int t = wv; t < 64; t += 8) {
    int kt = (t >> 4) * 16;
    int ct = (t & 15) * 16;
    v8f acc = v8f_zero();
    for (int n0 = 0; n0 < N; n0 += 32) {
      v16bf a  = load_frag(stb, N, kt, n0);
      v16bf bb = load_frag(xtb, N, ct, n0);
      acc = wmma_bf(a, bb, acc);
    }
    int c = ct + colb;
    for (int i = 0; i < 8; i++) {
      int k = kt + i + rsh;
      float v = acc[i] / (colsum[b * K_ + k] + EPS_);
      cfea_cat[((size_t)b * MC + lvloff + k) * C_ + c] = v;
      cfeaT_bf[((size_t)b * C_ + c) * K_ + k] = f2bf(v);
    }
  }
}

__global__ void k_cxyz(const float* s_f32, const float* xyz, const float* colsum,
                       float* cxyz_cat, int N, int lvloff) {
  int b = blockIdx.x / K_;
  int k = blockIdx.x % K_;
  const float* sb = s_f32 + (size_t)b * N * K_;
  const float* xb = xyz + (size_t)b * N * 3;
  __shared__ float red[256][3];
  float a0 = 0, a1 = 0, a2 = 0;
  for (int n = threadIdx.x; n < N; n += 256) {
    float sv = sb[(size_t)n * K_ + k];
    a0 += sv * xb[n*3]; a1 += sv * xb[n*3+1]; a2 += sv * xb[n*3+2];
  }
  red[threadIdx.x][0]=a0; red[threadIdx.x][1]=a1; red[threadIdx.x][2]=a2;
  __syncthreads();
  for (int s = 128; s > 0; s >>= 1) {
    if (threadIdx.x < s) {
      red[threadIdx.x][0]+=red[threadIdx.x+s][0];
      red[threadIdx.x][1]+=red[threadIdx.x+s][1];
      red[threadIdx.x][2]+=red[threadIdx.x+s][2];
    }
    __syncthreads();
  }
  if (threadIdx.x == 0) {
    float inv = 1.f / (colsum[b*K_+k] + EPS_);
    float* o = &cxyz_cat[((size_t)b * MC + lvloff + k) * 3];
    o[0]=red[0][0]*inv; o[1]=red[0][1]*inv; o[2]=red[0][2]*inv;
  }
}

// assignment + welsch losses for one level
__global__ void k_loss_ac(const float* s_f32, const float* xyz, const float* cxyz_cat,
                          int N, int lvloff, float scale, float* acc_ass, float* acc_wel) {
  int gid = blockIdx.x * 256 + threadIdx.x;
  int b = gid / N, n = gid % N;
  __shared__ float cx[K_][3];
  if (threadIdx.x < K_ * 3) {
    int k = threadIdx.x / 3, d = threadIdx.x % 3;
    cx[k][d] = cxyz_cat[((size_t)b * MC + lvloff + k) * 3 + d];
  }
  __syncthreads();
  const float* xp = &xyz[((size_t)b * N + n) * 3];
  float px = xp[0], py = xp[1], pz = xp[2];
  const float* sp = &s_f32[((size_t)b * N + n) * K_];
  float al = 0.f, dmin = 1e30f;
  for (int k = 0; k < K_; k++) {
    float dx = px - cx[k][0], dy = py - cx[k][1], dz = pz - cx[k][2];
    float d2 = dx*dx + dy*dy + dz*dz;
    al += sp[k] * d2;
    dmin = fminf(dmin, d2);
  }
  float wl = 1.f - __expf(-dmin * 0.5f);
  __shared__ float r1[256], r2[256];
  r1[threadIdx.x] = al; r2[threadIdx.x] = wl;
  __syncthreads();
  for (int s = 128; s > 0; s >>= 1) {
    if (threadIdx.x < s) { r1[threadIdx.x]+=r1[threadIdx.x+s]; r2[threadIdx.x]+=r2[threadIdx.x+s]; }
    __syncthreads();
  }
  if (threadIdx.x == 0) { atomicAdd(acc_ass, r1[0]*scale); atomicAdd(acc_wel, r2[0]*scale); }
}

// c2p = s @ c_fea; accumulate || l2norm(c2p) - nf ||_F^2
__global__ void k_c2p(const unsigned short* s_bf, const unsigned short* cfeaT_bf,
                      const float* nf_f32, int N, float* acc_c2p) {
  __shared__ float red[8][16], redb[8][16];
  int wv = threadIdx.x >> 5, lane = threadIdx.x & 31;
  int gw = blockIdx.x * 8 + wv;
  int rb = N >> 4;
  int b = gw / rb;
  int r0 = (gw % rb) << 4;
  const unsigned short* sb = s_bf + (size_t)b * N * K_;
  const unsigned short* ct = cfeaT_bf + (size_t)b * C_ * K_;
  const float* nfb = nf_f32 + (size_t)b * N * C_;
  int colb = lane & 15, rsh = (lane & 16) ? 8 : 0;
  v8f cacc[16];
  for (int f = 0; f < 16; f++) cacc[f] = v8f_zero();
  for (int kk = 0; kk < K_; kk += 32) {
    v16bf a = load_frag(sb, K_, r0, kk);
    for (int f = 0; f < 16; f++) {
      v16bf bb = load_frag(ct, K_, f * 16, kk);
      cacc[f] = wmma_bf(a, bb, cacc[f]);
    }
  }
  float s1[8], s2[8];
  for (int i = 0; i < 8; i++) { s1[i]=0.f; s2[i]=0.f; }
  for (int f = 0; f < 16; f++) {
    int c = f * 16 + colb;
    for (int i = 0; i < 8; i++) {
      int row = r0 + i + rsh;
      float v = cacc[f][i];
      s1[i] += v * v;
      s2[i] += v * nfb[(size_t)row * C_ + c];
    }
  }
  if (lane < 16) { red[wv][lane] = 0.f; redb[wv][lane] = 0.f; }
  for (int i = 0; i < 8; i++) {
    atomicAdd(&red[wv][i + rsh],  s1[i]);
    atomicAdd(&redb[wv][i + rsh], s2[i]);
  }
  __syncthreads();
  if (lane < 16) {
    float S1 = red[wv][lane], S2 = redb[wv][lane];
    float r = fmaxf(sqrtf(S1), 1e-12f);
    atomicAdd(acc_c2p, S1/(r*r) - 2.f*S2/r + 1.0f);  // + ||nf||^2 (unit rows)
  }
}

// generic 256-column GEMM: out[b,N,C_] = A[b,N,Kdim] @ Bt[C_,Kdim]^T
__global__ void k_gemm256(const unsigned short* a_bf, int lda,
                          const unsigned short* bt_bf, int ldb, int Kdim,
                          int N, float* out_f32, unsigned short* out_bf) {
  int wv = threadIdx.x >> 5, lane = threadIdx.x & 31;
  int gw = blockIdx.x * 8 + wv;
  int rb = N >> 4;
  int b = gw / rb;
  int r0 = (gw % rb) << 4;
  const unsigned short* ab = a_bf + (size_t)b * N * lda;
  int colb = lane & 15, rsh = (lane & 16) ? 8 : 0;
  v8f cacc[16];
  for (int f = 0; f < 16; f++) cacc[f] = v8f_zero();
  for (int kk = 0; kk < Kdim; kk += 32) {
    v16bf a = load_frag(ab, lda, r0, kk);
    for (int f = 0; f < 16; f++) {
      v16bf bb = load_frag(bt_bf, ldb, f * 16, kk);
      cacc[f] = wmma_bf(a, bb, cacc[f]);
    }
  }
  for (int f = 0; f < 16; f++) {
    int c = f * 16 + colb;
    for (int i = 0; i < 8; i++) {
      size_t o = ((size_t)b * N + r0 + i + rsh) * C_ + c;
      if (out_f32) out_f32[o] = cacc[f][i];
      if (out_bf)  out_bf[o]  = f2bf(cacc[f][i]);
    }
  }
}

// ---------------- global (128-node) stage: scalar kernels ----------------
__global__ void k_gknn(const float* cxyz_cat, float* na_c) {
  int b = blockIdx.x, i = threadIdx.x;       // 128 threads
  __shared__ float cx[MC][3];
  __shared__ float thr[MC];
  __shared__ float degl[MC];
  cx[i][0] = cxyz_cat[((size_t)b*MC+i)*3+0];
  cx[i][1] = cxyz_cat[((size_t)b*MC+i)*3+1];
  cx[i][2] = cxyz_cat[((size_t)b*MC+i)*3+2];
  __syncthreads();
  float best[8];
  for (int k = 0; k < 8; k++) best[k] = 1e30f;
  for (int j = 0; j < MC; j++) {
    float dx=cx[i][0]-cx[j][0], dy=cx[i][1]-cx[j][1], dz=cx[i][2]-cx[j][2];
    float v = dx*dx+dy*dy+dz*dz + (j==i ? 1e6f : 0.f);
    if (v < best[7]) {
      best[7] = v;
      for (int k = 7; k > 0; k--)
        if (best[k] < best[k-1]) { float t=best[k]; best[k]=best[k-1]; best[k-1]=t; }
    }
  }
  thr[i] = best[7];
  __syncthreads();
  float dg = 0.f;
  for (int j = 0; j < MC; j++) {
    if (j == i) continue;
    float dx=cx[i][0]-cx[j][0], dy=cx[i][1]-cx[j][1], dz=cx[i][2]-cx[j][2];
    float d2 = dx*dx+dy*dy+dz*dz;
    float e = __expf(-d2);
    float aij = (d2 <= thr[i]) ? e : 0.f;
    float aji = (d2 <= thr[j]) ? e : 0.f;
    dg += fmaxf(aij, aji);
  }
  degl[i] = dg;
  __syncthreads();
  float di = rsqrtf(degl[i] + EPS_);
  for (int j = 0; j < MC; j++) {
    float v = 0.f;
    if (j != i) {
      float dx=cx[i][0]-cx[j][0], dy=cx[i][1]-cx[j][1], dz=cx[i][2]-cx[j][2];
      float d2 = dx*dx+dy*dy+dz*dz;
      float e = __expf(-d2);
      float aij = (d2 <= thr[i]) ? e : 0.f;
      float aji = (d2 <= thr[j]) ? e : 0.f;
      v = fmaxf(aij, aji) * di * rsqrtf(degl[j] + EPS_);
    }
    na_c[((size_t)b*MC+i)*MC+j] = v;
  }
}

__global__ void k_gXc(const float* na_c, const float* cfea_cat, float* Xc) {
  int b = blockIdx.x;
  const float* na = na_c + (size_t)b*MC*MC;
  const float* cf = cfea_cat + (size_t)b*MC*C_;
  float* xc = Xc + (size_t)b*MC*C_;
  for (int o = threadIdx.x; o < MC*C_; o += blockDim.x) {
    int r = o / C_, c = o % C_;
    float s = 0.f;
    for (int j = 0; j < MC; j++) s += na[r*MC+j] * cf[j*C_+c];
    xc[o] = s;
  }
}

__global__ void k_gh(const float* Xc, const float* gW1, const float* gb1, float* h2) {
  int b = blockIdx.x;
  const float* xc = Xc + (size_t)b*MC*C_;
  float* h = h2 + (size_t)b*MC*H_;
  for (int o = threadIdx.x; o < MC*H_; o += blockDim.x) {
    int r = o / H_, c = o % H_;
    float s = gb1[c];
    for (int k = 0; k < C_; k++) s += xc[r*C_+k] * gW1[k*H_+c];
    h[o] = fmaxf(s, 0.f);
  }
}

__global__ void k_gs(const float* h2, const float* gW2, const float* gb2,
                     float* sg, float* sgcs) {
  int b = blockIdx.x, r = threadIdx.x;        // 128 threads
  __shared__ float sl[MC][KG_];
  const float* h = h2 + (size_t)b*MC*H_;
  float lg[KG_];
  float mx = -1e30f;
  for (int c = 0; c < KG_; c++) {
    float s = gb2[c];
    for (int k = 0; k < H_; k++) s += h[r*H_+k] * gW2[k*KG_+c];
    s *= (1.f / TAU_);
    lg[c] = s; mx = fmaxf(mx, s);
  }
  float sm = 0.f;
  for (int c = 0; c < KG_; c++) { lg[c] = __expf(lg[c]-mx); sm += lg[c]; }
  for (int c = 0; c < KG_; c++) {
    lg[c] /= sm;
    sl[r][c] = lg[c];
    sg[((size_t)b*MC+r)*KG_+c] = lg[c];
  }
  __syncthreads();
  if (r < KG_) {
    float cs = 0.f;
    for (int j = 0; j < MC; j++) cs += sl[j][r];
    sgcs[b*KG_+r] = cs;
  }
}

__global__ void k_gcf(const float* sg, const float* sgcs, const float* cfea_cat,
                      const float* cxyz_cat, float* cfg, float* cxyzg) {
  int b = blockIdx.x;
  for (int o = threadIdx.x; o < KG_*C_; o += blockDim.x) {
    int k = o / C_, c = o % C_;
    float s = 0.f;
    for (int j = 0; j < MC; j++)
      s += sg[((size_t)b*MC+j)*KG_+k] * cfea_cat[((size_t)b*MC+j)*C_+c];
    cfg[((size_t)b*KG_+k)*C_+c] = s / (sgcs[b*KG_+k] + EPS_);
  }
  if (threadIdx.x < KG_*3) {
    int k = threadIdx.x / 3, d = threadIdx.x % 3;
    float s = 0.f;
    for (int j = 0; j < MC; j++)
      s += sg[((size_t)b*MC+j)*KG_+k] * cxyz_cat[((size_t)b*MC+j)*3+d];
    cxyzg[((size_t)b*KG_+k)*3+d] = s / (sgcs[b*KG_+k] + EPS_);
  }
}

__global__ void k_gcut(const float* sg, const float* cxyz_cat, const float* cxyzg, float* acc) {
  int b = blockIdx.x, r = threadIdx.x;        // 128 threads
  __shared__ float red[MC];
  const float* p = &cxyz_cat[((size_t)b*MC+r)*3];
  float px = p[0], py = p[1], pz = p[2];
  float al = 0.f;
  for (int k = 0; k < KG_; k++) {
    const float* q = &cxyzg[((size_t)b*KG_+k)*3];
    float dx = px-q[0], dy = py-q[1], dz = pz-q[2];
    al += sg[((size_t)b*MC+r)*KG_+k] * (dx*dx+dy*dy+dz*dz);
  }
  red[r] = al;
  __syncthreads();
  for (int s = 64; s > 0; s >>= 1) { if (r < s) red[r] += red[r+s]; __syncthreads(); }
  if (r == 0) atomicAdd(acc, red[0] / (float)(B_*MC));
}

__global__ void k_gkv(const float* cfg, const float* Wk, const float* Wv,
                      float* kmat, float* vmat) {
  int b = blockIdx.x;
  for (int o = threadIdx.x; o < KG_*C_; o += blockDim.x) {
    int m = o / C_, c = o % C_;
    float sk = 0.f, sv = 0.f;
    for (int d = 0; d < C_; d++) {
      float f = cfg[((size_t)b*KG_+m)*C_+d];
      sk += f * Wk[d*C_+c];
      sv += f * Wv[d*C_+c];
    }
    kmat[((size_t)b*KG_+m)*C_+c] = sk;
    vmat[((size_t)b*KG_+m)*C_+c] = sv;
  }
}

__global__ void k_attn_core(const unsigned short* q_bf, const float* kmat, const float* vmat,
                            unsigned short* out_bf) {
  int gid = blockIdx.x * 256 + threadIdx.x;
  int b = gid / NSP;
  const unsigned short* q = q_bf + (size_t)gid * C_;
  const float* km = kmat + (size_t)b*KG_*C_;
  const float* vm = vmat + (size_t)b*KG_*C_;
  unsigned short* ob = out_bf + (size_t)gid * C_;
  const int DH = C_ / 4;
  for (int h = 0; h < 4; h++) {
    float sc[KG_];
    float mx = -1e30f;
    for (int m = 0; m < KG_; m++) {
      float s = 0.f;
      for (int d = 0; d < DH; d++) s += bf2f(q[h*DH+d]) * km[m*C_+h*DH+d];
      s *= 0.125f;            // 1/sqrt(64)
      sc[m] = s; mx = fmaxf(mx, s);
    }
    float sm = 0.f;
    for (int m = 0; m < KG_; m++) { sc[m] = __expf(sc[m]-mx); sm += sc[m]; }
    float inv = 1.f / sm;
    for (int d = 0; d < DH; d++) {
      float o = 0.f;
      for (int m = 0; m < KG_; m++) o += sc[m] * vm[m*C_+h*DH+d];
      ob[h*DH+d] = f2bf(o * inv);
    }
  }
}

// quadratic form x^T na x for the g_xyz cut loss (WMMA gram from LDS-staged panel)
__global__ void k_quad(const unsigned short* nf_bf, const float* xyz, const float* dinv,
                       float* qsum, int N, float r2) {
  __shared__ __align__(16) unsigned short btile[16 * C_];
  __shared__ float cxyzs[16][3];
  __shared__ float cdins[16];
  __shared__ float rxyz[8][16][3];
  __shared__ float rdin[8][16];
  __shared__ float redw[8];
  int wv = threadIdx.x >> 5, lane = threadIdx.x & 31;
  int gw = blockIdx.x * 8 + wv;
  int rb = N >> 4;
  int b = gw / rb;
  int r0 = (gw % rb) << 4;
  const unsigned short* nfb = nf_bf + (size_t)b * N * C_;
  const float* xb = xyz + (size_t)b * N * 3;
  const float* db = dinv + (size_t)b * N;
  if (lane < 16) {
    rxyz[wv][lane][0] = xb[(r0+lane)*3];
    rxyz[wv][lane][1] = xb[(r0+lane)*3+1];
    rxyz[wv][lane][2] = xb[(r0+lane)*3+2];
    rdin[wv][lane] = db[r0+lane];
  }
  if (lane == 0) redw[wv] = 0.f;
  v16bf afr[8];
#pragma unroll
  for (int j = 0; j < 8; j++) afr[j] = load_frag(nfb, C_, r0, j * 32);
  float part = 0.f;
  int colb = lane & 15, rsh = (lane & 16) ? 8 : 0;
  int srow = threadIdx.x >> 4, sseg = threadIdx.x & 15;
  for (int m0 = 0; m0 < N; m0 += 16) {
    __syncthreads();
    {
      const U4* src = (const U4*)(nfb + (size_t)(m0 + srow) * C_ + sseg * 16);
      U4* dst = (U4*)(&btile[srow * C_ + sseg * 16]);
      dst[0] = src[0];
      dst[1] = src[1];
    }
    if (threadIdx.x < 48)
      cxyzs[threadIdx.x / 3][threadIdx.x % 3] =
          xb[(size_t)(m0 + threadIdx.x / 3) * 3 + threadIdx.x % 3];
    if (threadIdx.x < 16) cdins[threadIdx.x] = db[m0 + threadIdx.x];
    __syncthreads();
    v8f acc = v8f_zero();
#pragma unroll
    for (int j = 0; j < 8; j++) {
      v16bf bb = load_frag(btile, C_, 0, j * 32);
      acc = wmma_bf(afr[j], bb, acc);
    }
    int col = m0 + colb;
    float cx = cxyzs[colb][0], cy = cxyzs[colb][1], cz = cxyzs[colb][2];
    float dc = cdins[colb];
    for (int i = 0; i < 8; i++) {
      int row = r0 + i + rsh;
      float rx = rxyz[wv][i+rsh][0], ry = rxyz[wv][i+rsh][1], rz = rxyz[wv][i+rsh][2];
      float dx = rx-cx, dy = ry-cy, dz = rz-cz;
      float d2x = dx*dx+dy*dy+dz*dz;
      float d2f = fmaxf(2.f - 2.f*acc[i], 0.f);
      float A = __expf(-GAMMA_*d2f/(float)C_);
      A = (d2x < r2 && row != col) ? A : 0.f;
      part += A * rdin[wv][i+rsh] * dc * (rx*cx + ry*cy + rz*cz);
    }
  }
  atomicAdd(&redw[wv], part);
  if (lane == 0) atomicAdd(&qsum[b], redw[wv]);
}

__global__ void k_sumxx(const float* xyz, float* sumxx, int N) {
  int b = blockIdx.x;
  __shared__ float red[256];
  float s = 0.f;
  for (int n = threadIdx.x; n < N; n += 256) {
    const float* p = &xyz[((size_t)b*N+n)*3];
    s += p[0]*p[0] + p[1]*p[1] + p[2]*p[2];
  }
  red[threadIdx.x] = s;
  __syncthreads();
  for (int t = 128; t > 0; t >>= 1) { if (threadIdx.x < t) red[threadIdx.x] += red[threadIdx.x+t]; __syncthreads(); }
  if (threadIdx.x == 0) sumxx[b] = red[0];
}

__global__ void k_gloss(const float* agg, const float* gfea, float* acc) {
  int row = blockIdx.x;                 // b*NSP+n
  const float* a = agg + (size_t)row * C_;
  const float* g = gfea + (size_t)row * C_;
  __shared__ float r1[256], r2[256], r3[256];
  float av = a[threadIdx.x], gv = g[threadIdx.x];
  r1[threadIdx.x] = av*av; r2[threadIdx.x] = gv*gv; r3[threadIdx.x] = av*gv;
  __syncthreads();
  for (int t = 128; t > 0; t >>= 1) {
    if (threadIdx.x < t) { r1[threadIdx.x]+=r1[threadIdx.x+t]; r2[threadIdx.x]+=r2[threadIdx.x+t]; r3[threadIdx.x]+=r3[threadIdx.x+t]; }
    __syncthreads();
  }
  if (threadIdx.x == 0) {
    float ra = fmaxf(sqrtf(r1[0]), 1e-12f), rg = fmaxf(sqrtf(r2[0]), 1e-12f);
    float d = r1[0]/(ra*ra) + r2[0]/(rg*rg) - 2.f*r3[0]/(ra*rg);
    atomicAdd(acc, sqrtf(fmaxf(d, 0.f)) / (float)B_);
  }
}

__global__ void k_final(const float* scal, float* out2) {
  if (threadIdx.x == 0 && blockIdx.x == 0) {
    // scal: 0 ass, 1 wel, 2 c2p0, 3 c2p1, 4 cut_loss_c, 5 gl, 8..9 qsum, 10..11 sumxx
    float l = scal[0] + scal[1] + sqrtf(fmaxf(scal[2],0.f)) + sqrtf(fmaxf(scal[3],0.f)) + scal[4];
    float ass_g = 0.f;
    for (int b = 0; b < B_; b++) ass_g += (scal[10+b] - scal[8+b]) / (float)NSP;
    ass_g = 1e-4f * ass_g / (float)B_;
    out2[0] = l;
    out2[1] = scal[5] + ass_g;
  }
}

// ---------------- host driver ----------------
extern "C" void kernel_launch(void* const* d_in, const int* in_sizes, int n_in,
                              void* d_out, int out_size, void* d_ws, size_t ws_size,
                              hipStream_t stream) {
  (void)in_sizes; (void)n_in; (void)out_size; (void)ws_size;
  const float* g_fea = (const float*)d_in[0];
  const float* g_xyz = (const float*)d_in[1];
  const float* pfea[2] = { (const float*)d_in[2], (const float*)d_in[4] };
  const float* pxyz[2] = { (const float*)d_in[3], (const float*)d_in[5] };
  const float* lW1 = (const float*)d_in[6];
  const float* lb1 = (const float*)d_in[7];
  const float* lW2 = (const float*)d_in[8];
  const float* lb2 = (const float*)d_in[9];
  const float* gW1 = (const float*)d_in[10];
  const float* gb1 = (const float*)d_in[11];
  const float* gW2 = (const float*)d_in[12];
  const float* gb2 = (const float*)d_in[13];
  const float* Wq  = (const float*)d_in[14];
  const float* Wk  = (const float*)d_in[15];
  const float* Wv  = (const float*)d_in[16];
  const float* Wo  = (const float*)d_in[17];
  float* agg = (float*)d_out;

  char* base = (char*)d_ws;
  size_t cur = 0;
  auto alloc = [&](size_t bytes) -> void* {
    void* p = base + cur;
    cur = (cur + bytes + 255) & ~((size_t)255);
    return p;
  };
  const int N0 = 4096;
  float*  scal     = (float*)alloc(16 * 4);           // accumulators
  float*  colsum   = (float*)alloc(B_ * K_ * 4);
  unsigned short* w1t     = (unsigned short*)alloc((size_t)H_ * C_ * 2);
  unsigned short* w2t     = (unsigned short*)alloc((size_t)K_ * H_ * 2);
  unsigned short* wqt     = (unsigned short*)alloc((size_t)C_ * C_ * 2);
  unsigned short* wot     = (unsigned short*)alloc((size_t)C_ * C_ * 2);
  unsigned short* gfea_bf = (unsigned short*)alloc((size_t)B_ * NSP * C_ * 2);
  unsigned short* nfg_bf  = (unsigned short*)alloc((size_t)B_ * NSP * C_ * 2);
  unsigned short* q_bf    = (unsigned short*)alloc((size_t)B_ * NSP * C_ * 2);
  unsigned short* attn_bf = (unsigned short*)alloc((size_t)B_ * NSP * C_ * 2);
  float*  deg_g    = (float*)alloc((size_t)B_ * NSP * 4);
  float*  dinv_g   = (float*)alloc((size_t)B_ * NSP * 4);
  float*  nf_f32   = (float*)alloc((size_t)B_ * N0 * C_ * 4);
  unsigned short* nf_bf   = (unsigned short*)alloc((size_t)B_ * N0 * C_ * 2);
  unsigned short* xt_bf   = (unsigned short*)alloc((size_t)B_ * N0 * C_ * 2);
  unsigned short* m_bf    = (unsigned short*)alloc((size_t)B_ * N0 * C_ * 2);
  float*  deg      = (float*)alloc((size_t)B_ * N0 * 4);
  float*  dinv     = (float*)alloc((size_t)B_ * N0 * 4);
  float*  s_f32    = (float*)alloc((size_t)B_ * N0 * K_ * 4);
  unsigned short* s_bf    = (unsigned short*)alloc((size_t)B_ * N0 * K_ * 2);
  unsigned short* st_bf   = (unsigned short*)alloc((size_t)B_ * N0 * K_ * 2);
  unsigned short* cfeaT   = (unsigned short*)alloc((size_t)B_ * C_ * K_ * 2);
  float*  cfea_cat = (float*)alloc((size_t)B_ * MC * C_ * 4);
  float*  cxyz_cat = (float*)alloc((size_t)B_ * MC * 3 * 4);
  float*  na_c     = (float*)alloc((size_t)B_ * MC * MC * 4);
  float*  Xc       = (float*)alloc((size_t)B_ * MC * C_ * 4);
  float*  h2       = (float*)alloc((size_t)B_ * MC * H_ * 4);
  float*  sg       = (float*)alloc((size_t)B_ * MC * KG_ * 4);
  float*  sgcs     = (float*)alloc((size_t)B_ * KG_ * 4);
  float*  cfg      = (float*)alloc((size_t)B_ * KG_ * C_ * 4);
  float*  cxyzg    = (float*)alloc((size_t)B_ * KG_ * 3 * 4);
  float*  kmat     = (float*)alloc((size_t)B_ * KG_ * C_ * 4);
  float*  vmat     = (float*)alloc((size_t)B_ * KG_ * C_ * 4);

  hipMemsetAsync(scal, 0, 16 * 4, stream);

  // weight prep (bf16, transposed to contraction-major)
  k_transpose_bf16<<<(C_*H_+255)/256, 256, 0, stream>>>(lW1, w1t, C_, H_);
  k_transpose_bf16<<<(H_*K_+255)/256, 256, 0, stream>>>(lW2, w2t, H_, K_);
  k_transpose_bf16<<<(C_*C_+255)/256, 256, 0, stream>>>(Wq, wqt, C_, C_);
  k_transpose_bf16<<<(C_*C_+255)/256, 256, 0, stream>>>(Wo, wot, C_, C_);

  // g_fea conversions (raw bf16 for attention A-operand; normalized bf16 for A_g gram)
  k_rownorm<<<B_*NSP, 256, 0, stream>>>(g_fea, NSP, nullptr, nfg_bf, gfea_bf, nullptr);

  // -------- per-level pipeline --------
  for (int lvl = 0; lvl < 2; lvl++) {
    int N = lvl ? 2048 : 4096;
    hipMemsetAsync(colsum, 0, B_ * K_ * 4, stream);
    k_rownorm<<<B_*N, 256, 0, stream>>>(pfea[lvl], N, nf_f32, nf_bf, nullptr, xt_bf);
    k_deg   <<<B_*N/128, 256, 0, stream>>>(nf_bf, pxyz[lvl], deg, N, 1.0f);
    k_rsqrt <<<(B_*N+255)/256, 256, 0, stream>>>(deg, dinv, B_*N);
    k_naX   <<<B_*N/128, 256, 0, stream>>>(nf_bf, xt_bf, pxyz[lvl], dinv, m_bf, N, 1.0f);
    k_mincut<<<B_*N/128, 256, 0, stream>>>(m_bf, w1t, lb1, w2t, lb2,
                                           s_f32, s_bf, st_bf, colsum, N);
    k_cfea  <<<B_, 256, 0, stream>>>(st_bf, xt_bf, colsum, cfea_cat, cfeaT, N, lvl*K_);
    k_cxyz  <<<B_*K_, 256, 0, stream>>>(s_f32, pxyz[lvl], colsum, cxyz_cat, N, lvl*K_);
    k_loss_ac<<<B_*N/256, 256, 0, stream>>>(s_f32, pxyz[lvl], cxyz_cat, N, lvl*K_,
                                            1.0f/(float)(B_*N), &scal[0], &scal[1]);
    k_c2p   <<<B_*N/128, 256, 0, stream>>>(s_bf, cfeaT, nf_f32, N, &scal[2+lvl]);
  }

  // -------- global 128-node stage --------
  k_gknn<<<B_, MC, 0, stream>>>(cxyz_cat, na_c);
  k_gXc <<<B_, 256, 0, stream>>>(na_c, cfea_cat, Xc);
  k_gh  <<<B_, 256, 0, stream>>>(Xc, gW1, gb1, h2);
  k_gs  <<<B_, MC, 0, stream>>>(h2, gW2, gb2, sg, sgcs);
  k_gcf <<<B_, 256, 0, stream>>>(sg, sgcs, cfea_cat, cxyz_cat, cfg, cxyzg);
  k_gcut<<<B_, MC, 0, stream>>>(sg, cxyz_cat, cxyzg, &scal[4]);

  // -------- cross attention --------
  k_gemm256<<<B_*NSP/128, 256, 0, stream>>>(gfea_bf, C_, wqt, C_, C_, NSP, nullptr, q_bf);
  k_gkv     <<<B_, 256, 0, stream>>>(cfg, Wk, Wv, kmat, vmat);
  k_attn_core<<<B_*NSP/256, 256, 0, stream>>>(q_bf, kmat, vmat, attn_bf);
  k_gemm256<<<B_*NSP/128, 256, 0, stream>>>(attn_bf, C_, wot, C_, C_, NSP, agg, nullptr);

  // -------- A_g cut loss + g_loss --------
  k_deg  <<<B_*NSP/128, 256, 0, stream>>>(nfg_bf, g_xyz, deg_g, NSP, 16.0f);
  k_rsqrt<<<(B_*NSP+255)/256, 256, 0, stream>>>(deg_g, dinv_g, B_*NSP);
  k_quad <<<B_*NSP/128, 256, 0, stream>>>(nfg_bf, g_xyz, dinv_g, &scal[8], NSP, 16.0f);
  k_sumxx<<<B_, 256, 0, stream>>>(g_xyz, &scal[10], NSP);
  k_gloss<<<B_*NSP, 256, 0, stream>>>(agg, g_fea, &scal[5]);

  k_final<<<1, 32, 0, stream>>>(scal, agg + (size_t)B_*NSP*C_);
}
